// GAT_2396591751676
// MI455X (gfx1250) — matrix operand order repaired and checked by reading, not compiled, not run on hardware
//
#include <hip/hip_runtime.h>
#include <hip/hip_bf16.h>
#include <math.h>

// ---------------------------------------------------------------------------
// GAT (2-layer, H=4, C=32) + global mean/max pool + linear head for MI455X.
// GEMMs run on the bf16 WMMA pipe (v_wmma_f32_16x16x32_bf16, f32 accum).
// Weight tiles stream to LDS via CDNA5 async global->LDS loads (ASYNCcnt).
// Edge aggregation is wave-per-edge, coalesced gathers + L2-resident atomics.
// ---------------------------------------------------------------------------

#define F_IN   159
#define KPAD1  160          // F_IN padded to multiple of 32
#define HC     128          // H * C
#define NHEAD  4
#define GNUM   32

typedef __attribute__((ext_vector_type(16))) __bf16 v16bf;
typedef __attribute__((ext_vector_type(8)))  float  v8f;

#if defined(__has_builtin)
#if __has_builtin(__builtin_amdgcn_global_load_async_to_lds_b128) && \
    __has_builtin(__builtin_amdgcn_s_wait_asynccnt)
#define USE_ASYNC_LDS 1
#endif
#endif
#ifndef USE_ASYNC_LDS
#define USE_ASYNC_LDS 0
#endif

#if USE_ASYNC_LDS
// Builtin signature (from compiler diagnostic): first arg is AS1 pointer to
// 'int __attribute__((vector_size(16)))'; second is the AS3 analogue.
typedef int v4i_b128 __attribute__((vector_size(16)));
typedef __attribute__((address_space(1))) v4i_b128 g_v4i;
typedef __attribute__((address_space(3))) v4i_b128 l_v4i;
// Flat address of a __shared__ object: low 32 bits == LDS byte offset
// (the LDS aperture keeps the low 32 address bits).  Global: flat == AS1 numerically.
__device__ __forceinline__ l_v4i* to_lds(void* p) {
    return (l_v4i*)(unsigned int)(unsigned long long)p;
}
__device__ __forceinline__ g_v4i* to_glb(const void* p) {
    return (g_v4i*)(unsigned long long)p;
}
#endif

// ---------------- utility ----------------
__device__ __forceinline__ void atomicMaxF(float* addr, float val) {
    unsigned int* ua = (unsigned int*)addr;
    unsigned int old = *ua;
    while (__uint_as_float(old) < val) {
        unsigned int assumed = old;
        old = atomicCAS(ua, assumed, __float_as_uint(val));
        if (old == assumed) break;
    }
}

__global__ void fill_kernel(float* __restrict__ p, float v, int n) {
    for (int i = blockIdx.x * blockDim.x + threadIdx.x; i < n;
         i += gridDim.x * blockDim.x)
        p[i] = v;
}

// Convert weights to bf16, pre-swizzled per 32-wide K-block:
//   Wsw[((kt*128 + col)*32) + kk] = W[(kt*32+kk)*128 + col]   (zero-padded K)
// so each GEMM K-step's B tile is one contiguous 8 KB block.
__global__ void convw_kernel(const float* __restrict__ W1,
                             const float* __restrict__ W2,
                             __bf16* __restrict__ W1s,
                             __bf16* __restrict__ W2s) {
    int tid = blockIdx.x * blockDim.x + threadIdx.x;
    const int n1 = KPAD1 * HC;
    if (tid < n1) {
        const int kk  = tid & 31;
        const int col = (tid >> 5) & 127;
        const int kt  = tid >> 12;
        const int k   = kt * 32 + kk;
        W1s[tid] = (__bf16)((k < F_IN) ? W1[k * HC + col] : 0.0f);
    } else if (tid < n1 + HC * HC) {
        const int t2  = tid - n1;
        const int kk  = t2 & 31;
        const int col = (t2 >> 5) & 127;
        const int kt  = t2 >> 12;
        W2s[t2] = (__bf16)W2[(kt * 32 + kk) * HC + col];
    }
}

// ---------------- bf16 WMMA GEMM:  H[M x 128] = A[M x Ka] * W[Kpad x 128] ---
// Block: 256 threads = 8 waves; 128x128 output tile; wave w owns rows w*16..+15.
// W is the pre-swizzled bf16 weight ([kt][col][kk] contiguous per K-block).
__global__ __launch_bounds__(256) void gemm_bf16_kernel(
    const float* __restrict__ A, int lda, int Ka,
    const __bf16* __restrict__ W, int nkt,
    float* __restrict__ Hout, int M)
{
    __shared__ __bf16 sA[128][32];      // [row][k]    8 KB
    __shared__ __bf16 sB[128][40];      // [col][k]    10 KB; stride 80 B (16B-aligned chunks)

    const int tid  = threadIdx.x;
    const int row0 = blockIdx.x * 128;
    const int w    = tid >> 5;
    const int lane = tid & 31;
    const int half = lane >> 4;
    const int mn   = lane & 15;

    v8f acc[8];
    #pragma unroll
    for (int t = 0; t < 8; ++t)
        #pragma unroll
        for (int i = 0; i < 8; ++i) acc[t][i] = 0.0f;

    for (int kt = 0; kt < nkt; ++kt) {
        const int k0 = kt * 32;
        const __bf16* gblk = W + (size_t)kt * (HC * 32);   // contiguous 8 KB

        // ---- stage B: 512 x 16B chunks, 2 per thread ----------------------
        // chunk q: col = q>>2, j = q&3  ->  sB[col][j*8 .. j*8+7]
#if USE_ASYNC_LDS
        {
            const int q0 = tid * 2;
            #pragma unroll
            for (int u = 0; u < 2; ++u) {
                const int q   = q0 + u;
                const int col = q >> 2;
                const int j   = q & 3;
                __builtin_amdgcn_global_load_async_to_lds_b128(
                    to_glb(gblk + q * 8), to_lds(&sB[col][j * 8]), 0, 0);
            }
        }
#else
        {
            const int q0 = tid * 2;
            #pragma unroll
            for (int u = 0; u < 2; ++u) {
                const int q   = q0 + u;
                const int col = q >> 2;
                const int j   = q & 3;
                #pragma unroll
                for (int i = 0; i < 8; ++i)
                    sB[col][j * 8 + i] = gblk[q * 8 + i];
            }
        }
#endif
        // ---- stage A tile (f32 -> bf16), 4096 elems / 256 thr = 16 each ---
        // Hoisted bounds check: unpredicated fast loop except on the rare
        // row/K tail (last block row; last K-block of layer 1).
        {
            const int eb = tid * 16;
            const int r  = eb >> 5;
            const int kk = eb & 31;
            const int gr = row0 + r;
            const float* ap = A + (size_t)gr * lda + (k0 + kk);
            if (gr < M && (k0 + kk + 15) < Ka) {
                #pragma unroll
                for (int i = 0; i < 16; ++i)
                    sA[r][kk + i] = (__bf16)ap[i];
            } else {
                #pragma unroll
                for (int i = 0; i < 16; ++i) {
                    const int k = k0 + kk + i;
                    float v = 0.0f;
                    if (gr < M && k < Ka) v = ap[i];
                    sA[r][kk + i] = (__bf16)v;
                }
            }
        }
#if USE_ASYNC_LDS
        __builtin_amdgcn_s_wait_asynccnt(0);
#endif
        __syncthreads();

        // ---- A fragment: 16-bit A 16x32 layout (lane m=0..15, halves split K)
        v16bf afr;
        const int arow = w * 16 + mn;
        #pragma unroll
        for (int q = 0; q < 8; ++q) {
            const int kb = (q < 4) ? (2 * q + 8 * half)
                                   : (16 + 2 * (q - 4) + 8 * half);
            afr[2 * q]     = sA[arow][kb];
            afr[2 * q + 1] = sA[arow][kb + 1];
        }
        // ---- 8 WMMA tiles across N ----------------------------------------
        #pragma unroll
        for (int t = 0; t < 8; ++t) {
            v16bf bfr;
            const int col = t * 16 + mn;
            #pragma unroll
            for (int q = 0; q < 8; ++q) {
                const int kb = (q < 4) ? (2 * q + 8 * half)
                                       : (16 + 2 * (q - 4) + 8 * half);
                bfr[2 * q]     = sB[col][kb];
                bfr[2 * q + 1] = sB[col][kb + 1];
            }
            acc[t] = __builtin_amdgcn_wmma_f32_16x16x32_bf16(
                false, afr, false, bfr, (short)0, acc[t], false, false);
        }
        __syncthreads();
    }

    // ---- epilogue: D layout — lane holds col mn, rows (8*half + r) ---------
    const int rbase = row0 + w * 16 + half * 8;
    #pragma unroll
    for (int t = 0; t < 8; ++t) {
        #pragma unroll
        for (int r = 0; r < 8; ++r) {
            const int row = rbase + r;
            if (row < M)
                Hout[(size_t)row * HC + t * 16 + mn] = acc[t][r];
        }
    }
}

// ---------------- per-node attention logits ---------------------------------
__global__ void attdot_kernel(const float* __restrict__ Hbuf,
                              const float* __restrict__ asrc,
                              const float* __restrict__ adst,
                              float* __restrict__ als, float* __restrict__ ald,
                              int N)
{
    int tid = blockIdx.x * blockDim.x + threadIdx.x;
    if (tid >= N * NHEAD) return;
    const int n = tid >> 2, h = tid & 3;
    const float* hr = Hbuf + (size_t)n * HC + h * 32;
    const float* as_ = asrc + h * 32;
    const float* ad_ = adst + h * 32;
    float s0 = 0.0f, s1 = 0.0f;
    #pragma unroll
    for (int c = 0; c < 32; ++c) {
        const float v = hr[c];
        s0 += v * as_[c];
        s1 += v * ad_[c];
    }
    als[tid] = s0;
    ald[tid] = s1;
}

// ---------------- edge pass 1: segment max of leaky_relu logits -------------
// Also warms L2 with h[src] rows (global_prefetch) for the gather in pass 2.
__global__ void edgemax_kernel(const int* __restrict__ ei, int E, int N,
                               const float* __restrict__ als,
                               const float* __restrict__ ald,
                               const float* __restrict__ Hbuf,
                               float* __restrict__ mbuf)
{
    int tid = blockIdx.x * blockDim.x + threadIdx.x;
    const int tot = (E + N) * NHEAD;
    if (tid >= tot) return;
    const int e = tid >> 2, h = tid & 3;
    int src, dst;
    if (e < E) { src = ei[e]; dst = ei[E + e]; }
    else       { src = e - E; dst = src; }          // self loop
    if (h == 0)
        __builtin_prefetch(Hbuf + (size_t)src * HC, 0, 1);  // -> global_prefetch
    float v = als[src * NHEAD + h] + ald[dst * NHEAD + h];
    v = (v > 0.0f) ? v : 0.2f * v;                  // leaky_relu(0.2)
    atomicMaxF(&mbuf[dst * NHEAD + h], v);
}

// ---------------- edge pass 2: exp, denom sum, weighted scatter-add ---------
// One wave32 per edge; lane = channel within each of the 4 heads.
__global__ void edgeagg_kernel(const int* __restrict__ ei, int E, int N,
                               const float* __restrict__ als,
                               const float* __restrict__ ald,
                               const float* __restrict__ mbuf,
                               float* __restrict__ sbuf,
                               const float* __restrict__ Hbuf,
                               float* __restrict__ Obuf)
{
    const int gwid = (blockIdx.x * blockDim.x + threadIdx.x) >> 5;
    const int lane = threadIdx.x & 31;
    if (gwid >= E + N) return;
    int src, dst;
    if (gwid < E) { src = ei[gwid]; dst = ei[E + gwid]; }
    else          { src = gwid - E; dst = src; }

    float ev = 0.0f;
    if (lane < NHEAD) {
        float v = als[src * NHEAD + lane] + ald[dst * NHEAD + lane];
        v = (v > 0.0f) ? v : 0.2f * v;
        ev = __expf(v - mbuf[dst * NHEAD + lane]);
        atomicAdd(&sbuf[dst * NHEAD + lane], ev);
    }
    const float e0 = __shfl(ev, 0, 32);
    const float e1 = __shfl(ev, 1, 32);
    const float e2 = __shfl(ev, 2, 32);
    const float e3 = __shfl(ev, 3, 32);

    const float* hs = Hbuf + (size_t)src * HC;
    float*       od = Obuf + (size_t)dst * HC;
    atomicAdd(&od[lane],      e0 * hs[lane]);
    atomicAdd(&od[32 + lane], e1 * hs[32 + lane]);
    atomicAdd(&od[64 + lane], e2 * hs[64 + lane]);
    atomicAdd(&od[96 + lane], e3 * hs[96 + lane]);
}

// ---------------- per-node: normalize by softmax denom, + bias, (ELU) -------
__global__ void finalize_kernel(float* __restrict__ out,
                                const float* __restrict__ sbuf,
                                const float* __restrict__ bias,
                                int N, int do_elu)
{
    int tid = blockIdx.x * blockDim.x + threadIdx.x;
    if (tid >= N * HC) return;
    const int n = tid >> 7, c = tid & 127, h = c >> 5;
    float v = out[tid] / (sbuf[n * NHEAD + h] + 1e-16f) + bias[c];
    if (do_elu) v = (v > 0.0f) ? v : (__expf(v) - 1.0f);
    out[tid] = v;
}

// ---------------- pooling: per-block run-length reduce, sorted batch_vec ----
__global__ __launch_bounds__(128) void pool_kernel(
    const float* __restrict__ act, const int* __restrict__ batch,
    float* __restrict__ psum, float* __restrict__ pmax,
    float* __restrict__ pcnt, int N)
{
    const int c  = threadIdx.x;               // channel 0..127
    const int n0 = blockIdx.x * 128;
    int cur = -1, run = 0;
    float lsum = 0.0f, lmax = -3.402823466e38f;
    for (int i = 0; i < 128; ++i) {
        const int n = n0 + i;
        if (n >= N) break;
        const int g = batch[n];
        if (g != cur) {
            if (cur >= 0) {
                atomicAdd(&psum[cur * HC + c], lsum);
                atomicMaxF(&pmax[cur * HC + c], lmax);
                if (c == 0) atomicAdd(&pcnt[cur], (float)run);
            }
            cur = g; run = 0; lsum = 0.0f; lmax = -3.402823466e38f;
        }
        const float v = act[(size_t)n * HC + c];
        lsum += v;
        lmax = fmaxf(lmax, v);
        ++run;
    }
    if (cur >= 0) {
        atomicAdd(&psum[cur * HC + c], lsum);
        atomicMaxF(&pmax[cur * HC + c], lmax);
        if (c == 0) atomicAdd(&pcnt[cur], (float)run);
    }
}

// ---------------- head: (mean + max) @ lin_w + lin_b ------------------------
__global__ __launch_bounds__(128) void pred_kernel(
    const float* __restrict__ psum, const float* __restrict__ pmax,
    const float* __restrict__ pcnt, const float* __restrict__ lin_w,
    const float* __restrict__ lin_b, float* __restrict__ out)
{
    const int g = blockIdx.x, c = threadIdx.x;
    float v = psum[g * HC + c] / pcnt[g] + pmax[g * HC + c];
    float p = v * lin_w[c];
    #pragma unroll
    for (int off = 16; off; off >>= 1) p += __shfl_down(p, off, 32);
    __shared__ float red[4];
    if ((c & 31) == 0) red[c >> 5] = p;
    __syncthreads();
    if (c == 0) out[g] = red[0] + red[1] + red[2] + red[3] + lin_b[0];
}

// ---------------------------------------------------------------------------
extern "C" void kernel_launch(void* const* d_in, const int* in_sizes, int n_in,
                              void* d_out, int out_size, void* d_ws, size_t ws_size,
                              hipStream_t stream) {
    const float* x      = (const float*)d_in[0];
    const int*   ei     = (const int*)d_in[1];
    const int*   batch  = (const int*)d_in[2];
    const float* W1     = (const float*)d_in[3];
    const float* asrc1  = (const float*)d_in[4];
    const float* adst1  = (const float*)d_in[5];
    const float* b1     = (const float*)d_in[6];
    const float* W2     = (const float*)d_in[7];
    const float* asrc2  = (const float*)d_in[8];
    const float* adst2  = (const float*)d_in[9];
    const float* b2     = (const float*)d_in[10];
    const float* lin_w  = (const float*)d_in[11];
    const float* lin_b  = (const float*)d_in[12];

    const int N  = in_sizes[0] / F_IN;
    const int E  = in_sizes[1] / 2;
    const int ET = E + N;

    // ---- workspace partition (256B aligned) ----
    char* wp = (char*)d_ws;
    auto alloc = [&](size_t bytes) -> void* {
        void* p = (void*)wp;
        wp += (bytes + 255) & ~(size_t)255;
        return p;
    };
    float*  buf0 = (float*)alloc((size_t)N * HC * 4);   // GEMM output h
    float*  buf1 = (float*)alloc((size_t)N * HC * 4);   // layer1 aggregate / layer2 GEMM input
    float*  buf2 = (float*)alloc((size_t)N * HC * 4);   // layer2 aggregate
    float*  als  = (float*)alloc((size_t)N * NHEAD * 4);
    float*  ald  = (float*)alloc((size_t)N * NHEAD * 4);
    float*  mb   = (float*)alloc((size_t)N * NHEAD * 4);
    float*  sb   = (float*)alloc((size_t)N * NHEAD * 4);
    __bf16* W1b  = (__bf16*)alloc((size_t)KPAD1 * HC * 2);
    __bf16* W2b  = (__bf16*)alloc((size_t)HC * HC * 2);
    float*  psum = (float*)alloc((size_t)GNUM * HC * 4);
    float*  pmax = (float*)alloc((size_t)GNUM * HC * 4);
    float*  pcnt = (float*)alloc((size_t)GNUM * 4);

    const int TB = 256;
    const int nNH  = N * NHEAD;
    const int nNC  = N * HC;
    const int nET4 = ET * NHEAD;
    const int gemmBlocks = (N + 127) / 128;
    const int aggBlocks  = (ET * 32 + TB - 1) / TB;
    const float NEG = -3.402823466e38f;

    // weights -> bf16 (pre-swizzled per K-block)
    convw_kernel<<<(KPAD1 * HC + HC * HC + TB - 1) / TB, TB, 0, stream>>>(W1, W2, W1b, W2b);

    // ================= layer 1 =================
    gemm_bf16_kernel<<<gemmBlocks, TB, 0, stream>>>(x, F_IN, F_IN, W1b, KPAD1 / 32, buf0, N);
    attdot_kernel<<<(nNH + TB - 1) / TB, TB, 0, stream>>>(buf0, asrc1, adst1, als, ald, N);
    fill_kernel<<<(nNH + TB - 1) / TB, TB, 0, stream>>>(mb, NEG, nNH);
    fill_kernel<<<(nNH + TB - 1) / TB, TB, 0, stream>>>(sb, 0.0f, nNH);
    fill_kernel<<<(nNC + TB - 1) / TB, TB, 0, stream>>>(buf1, 0.0f, nNC);
    edgemax_kernel<<<(nET4 + TB - 1) / TB, TB, 0, stream>>>(ei, E, N, als, ald, buf0, mb);
    edgeagg_kernel<<<aggBlocks, TB, 0, stream>>>(ei, E, N, als, ald, mb, sb, buf0, buf1);
    finalize_kernel<<<(nNC + TB - 1) / TB, TB, 0, stream>>>(buf1, sb, b1, N, 1);

    // ================= layer 2 =================
    gemm_bf16_kernel<<<gemmBlocks, TB, 0, stream>>>(buf1, HC, HC, W2b, HC / 32, buf0, N);
    attdot_kernel<<<(nNH + TB - 1) / TB, TB, 0, stream>>>(buf0, asrc2, adst2, als, ald, N);
    fill_kernel<<<(nNH + TB - 1) / TB, TB, 0, stream>>>(mb, NEG, nNH);
    fill_kernel<<<(nNH + TB - 1) / TB, TB, 0, stream>>>(sb, 0.0f, nNH);
    fill_kernel<<<(nNC + TB - 1) / TB, TB, 0, stream>>>(buf2, 0.0f, nNC);
    edgemax_kernel<<<(nET4 + TB - 1) / TB, TB, 0, stream>>>(ei, E, N, als, ald, buf0, mb);
    edgeagg_kernel<<<aggBlocks, TB, 0, stream>>>(ei, E, N, als, ald, mb, sb, buf0, buf2);
    finalize_kernel<<<(nNC + TB - 1) / TB, TB, 0, stream>>>(buf2, sb, b2, N, 0);

    // ================= pooling + head =================
    fill_kernel<<<(GNUM * HC + TB - 1) / TB, TB, 0, stream>>>(psum, 0.0f, GNUM * HC);
    fill_kernel<<<(GNUM * HC + TB - 1) / TB, TB, 0, stream>>>(pmax, NEG, GNUM * HC);
    fill_kernel<<<1, TB, 0, stream>>>(pcnt, 0.0f, GNUM);
    pool_kernel<<<(N + 127) / 128, 128, 0, stream>>>(buf2, batch, psum, pmax, pcnt, N);
    pred_kernel<<<GNUM, 128, 0, stream>>>(psum, pmax, pcnt, lin_w, lin_b, (float*)d_out);
}
